// GPT_87359634801307
// MI455X (gfx1250) — matrix operand order repaired and checked by reading, not compiled, not run on hardware
//
#include <hip/hip_runtime.h>
#include <cmath>

typedef __attribute__((ext_vector_type(16))) _Float16 v16h;
typedef __attribute__((ext_vector_type(8)))  float    v8f;
typedef _Float16 half_t;

constexpr int VOC = 50257;
constexpr int NL  = 6;
constexpr int DM  = 768;
constexpr int NH  = 12;
constexpr int HDI = 64;      // head dim
constexpr int BB  = 4;
constexpr int TT  = 512;
constexpr int BT  = BB * TT; // 2048
constexpr int FF  = 4 * DM;  // 3072

constexpr int TM = 256, TN = 128, TK = 32;
constexpr int LSTR = TK + 8;       // LDS row stride in halves (80B: 16B aligned + bank skew)
constexpr int ATILE_H = TM * LSTR; // A tile halves
constexpr int BTILE_H = TN * LSTR; // B tile halves

enum { GF_RELU = 1, GF_BIAS = 2, GF_RES = 4, GF_OUT32 = 8, GF_OUT16 = 16, GF_OUT16T = 32 };

// ---------------------------------------------------------------------------
// Async-copy a ROWSx32-half tile (f16 global, row-major) into LDS.
// ROWS*4 16B chunks spread over 256 threads (NCH chunks per thread), so each
// wave issues exactly NCH A/B async instructions -> uniform ASYNCcnt.
// Rows are CLAMPED (not skipped) so the issue count never diverges.
// ---------------------------------------------------------------------------
template <int ROWS, int NCH>
__device__ __forceinline__ void async_copy_tile(const half_t* __restrict__ gbase,
                                                int ld, int rowClamp,
                                                int tileRow0, int k0,
                                                half_t* lds, int tid) {
    static_assert(ROWS * 4 == NCH * 256, "chunk math");
#pragma unroll
    for (int j = 0; j < NCH; ++j) {
        const int g   = j * 256 + tid;  // chunk id, consecutive lanes -> consecutive 16B
        const int row = g >> 2;
        const int cir = g & 3;          // 16B chunk within 64B row
        int gr = tileRow0 + row;
        gr = gr < rowClamp ? gr : rowClamp;
        const unsigned long long ga =
            (unsigned long long)(const void*)(gbase + (long)gr * ld + k0 + cir * 8);
        const unsigned la =
            (unsigned)(unsigned long long)(const void*)(lds + row * LSTR + cir * 8);
        asm volatile("global_load_async_to_lds_b128 %0, %1, off"
                     :: "v"(la), "v"(ga) : "memory");
    }
}

// ---------------------------------------------------------------------------
// Batched WMMA GEMM: C[z] = act(A[z] @ W[z] + bias (+Rsd))
//   A : f16 [M,K] row-major (ldA)
//   W : f16 [N,K] row-major (N-major "pre-transposed" B operand, ldW)
//   out: f32 C32 and/or f16 C16 (optionally transposed [N,M])
// 256x128x32 tiles, 8 wave32s in 4(M)x2(N); each wave owns 64x64 = 16 WMMA
// accumulators (1.0 ds_load_b128 per WMMA). Double-buffered LDS fed by
// global_load_async_to_lds_b128 + s_wait_asynccnt pipelining.
// ---------------------------------------------------------------------------
__global__ __launch_bounds__(256) void wmma_gemm(
    const half_t* __restrict__ A, long sAz, int aDiv, int ldA,
    const half_t* __restrict__ W, long sWz, int wMod, int ldW,
    const float* __restrict__ bias, long sBz, int bMod,
    const float* __restrict__ Rsd, int ldR,
    float* __restrict__ C32, long sC32z, int ldC32,
    half_t* __restrict__ C16, long sC16z, int ldC16,
    int M, int N, int K, int flags)
{
    __shared__ alignas(16) half_t As[2 * ATILE_H];   // 40 KB
    __shared__ alignas(16) half_t Bs[2 * BTILE_H];   // 20 KB

    const int z = blockIdx.z;
    const half_t* Az = A + (long)(z / aDiv) * sAz;
    const half_t* Wz = W + (long)(z % wMod) * sWz;
    const float*  Bz = bias ? (bias + (long)(z % bMod) * sBz) : nullptr;

    const int tid  = threadIdx.x;
    const int lane = tid & 31;
    const int wave = tid >> 5;
    const int waveM = wave >> 1;  // 0..3
    const int waveN = wave & 1;   // 0..1

    const int m0 = blockIdx.y * TM;
    const int n0 = blockIdx.x * TN;

    v8f acc[4][4];
#pragma unroll
    for (int i = 0; i < 4; ++i)
#pragma unroll
        for (int j = 0; j < 4; ++j)
#pragma unroll
            for (int e = 0; e < 8; ++e) acc[i][j][e] = 0.0f;

    const int nk = K / TK;          // K is always a multiple of 32 here

    // prologue: stage tile 0 into buffer 0  (6 async instr per wave)
    async_copy_tile<TM, 4>(Az, ldA, M - 1, m0, 0, &As[0], tid);
    async_copy_tile<TN, 2>(Wz, ldW, N - 1, n0, 0, &Bs[0], tid);

    const int r15 = lane & 15;
    const int kbA = (lane >> 4) * 8;   // A: lane half selects K {0-7,16-23} vs {8-15,24-31}
    const int kbB = (lane >> 4) * 8;   // B: lane half selects K {0-7,16-23} vs {8-15,24-31}

    for (int kt = 0; kt < nk; ++kt) {
        const int  cur  = kt & 1;
        const bool more = (kt + 1) < nk;
        if (more) {
            // prefetch next tile into the other buffer, then wait for current
            async_copy_tile<TM, 4>(Az, ldA, M - 1, m0, (kt + 1) * TK,
                                   &As[(1 - cur) * ATILE_H], tid);
            async_copy_tile<TN, 2>(Wz, ldW, N - 1, n0, (kt + 1) * TK,
                                   &Bs[(1 - cur) * BTILE_H], tid);
            asm volatile("s_wait_asynccnt 6" ::: "memory");
        } else {
            asm volatile("s_wait_asynccnt 0" ::: "memory");
        }
        __syncthreads();

        const half_t* Asb = &As[cur * ATILE_H];
        const half_t* Bsb = &Bs[cur * BTILE_H];

        v16h bfrag[4];
#pragma unroll
        for (int nt = 0; nt < 4; ++nt) {
            const half_t* p = &Bsb[(waveN * 64 + nt * 16 + r15) * LSTR + kbB];
            v16h f;
            ((float4*)&f)[0] = *(const float4*)(p);
            ((float4*)&f)[1] = *(const float4*)(p + 16);
            bfrag[nt] = f;
        }
#pragma unroll
        for (int mt = 0; mt < 4; ++mt) {
            const half_t* p = &Asb[(waveM * 64 + mt * 16 + r15) * LSTR];
            v16h a;
            ((float4*)&a)[0] = *(const float4*)(p + kbA);
            ((float4*)&a)[1] = *(const float4*)(p + 16 + kbA);
#pragma unroll
            for (int nt = 0; nt < 4; ++nt) {
                acc[mt][nt] = __builtin_amdgcn_wmma_f32_16x16x32_f16(
                    false, a, false, bfrag[nt], (short)0, acc[mt][nt], false, false);
            }
        }
        __syncthreads();
    }

    // epilogue (C layout: lane 0-15 -> N=lane,M=r; lane 16-31 -> M=8+r)
    float*  C32z = (flags & GF_OUT32) ? (C32 + (long)z * sC32z) : nullptr;
    half_t* C16z = (flags & (GF_OUT16 | GF_OUT16T)) ? (C16 + (long)z * sC16z) : nullptr;
#pragma unroll
    for (int mt = 0; mt < 4; ++mt) {
        const int mb = m0 + waveM * 64 + mt * 16 + ((lane >> 4) * 8);
#pragma unroll
        for (int nt = 0; nt < 4; ++nt) {
            const int n = n0 + waveN * 64 + nt * 16 + (lane & 15);
            if (n >= N) continue;
            const float bv = (flags & GF_BIAS) ? Bz[n] : 0.0f;
#pragma unroll
            for (int r = 0; r < 8; ++r) {
                const int m = mb + r;
                if (m >= M) continue;
                float v = acc[mt][nt][r] + bv;
                if (flags & GF_RES)  v += Rsd[(long)m * ldR + n];
                if (flags & GF_RELU) v = v > 0.0f ? v : 0.0f;
                if (flags & GF_OUT32)  C32z[(long)m * ldC32 + n] = v;
                if (flags & GF_OUT16)  C16z[(long)m * ldC16 + n] = (half_t)v;
                if (flags & GF_OUT16T) C16z[(long)n * ldC16 + m] = (half_t)v;
            }
        }
    }
}

// ---------------------------------------------------------------------------
// Reductions + elementwise kernels
// ---------------------------------------------------------------------------
__device__ inline float blk_red(float v, float* sh, bool domax) {
#pragma unroll
    for (int o = 16; o > 0; o >>= 1) {
        float t = __shfl_down(v, o, 32);
        v = domax ? fmaxf(v, t) : v + t;
    }
    const int lane = threadIdx.x & 31, w = threadIdx.x >> 5;
    if (lane == 0) sh[w] = v;
    __syncthreads();
    if (threadIdx.x == 0) {
        float r = sh[0];
        const int nw = blockDim.x >> 5;
        for (int i = 1; i < nw; ++i) r = domax ? fmaxf(r, sh[i]) : r + sh[i];
        sh[0] = r;
    }
    __syncthreads();
    float r = sh[0];
    __syncthreads();
    return r;
}

__global__ void embed_k(const int* __restrict__ ids, const float* __restrict__ tok,
                        const float* __restrict__ pos, float* __restrict__ x) {
    const int i = blockIdx.x * blockDim.x + threadIdx.x;
    if (i >= BT * DM) return;
    const int d = i % DM, bt = i / DM, t = bt % TT;
    x[i] = tok[(long)ids[bt] * DM + d] + pos[t * DM + d];
}

// batched transposing f32->f16 convert: src [Z][K][N] -> dst [Z][N][K]
__global__ void cvtT16_k(const float* __restrict__ s, half_t* __restrict__ d,
                         int Kd, int Nd, long total) {
    const long i = (long)blockIdx.x * blockDim.x + threadIdx.x;
    if (i >= total) return;
    const long kn = (long)Kd * Nd;
    const long z = i / kn, r = i - z * kn;
    const int k = (int)(r / Nd), n = (int)(r - (long)k * Nd);
    d[z * kn + (long)n * Kd + k] = (half_t)s[i];
}

__global__ __launch_bounds__(256) void ln_k(const float* __restrict__ x,
                                            const float* __restrict__ g,
                                            const float* __restrict__ b,
                                            half_t* __restrict__ o) {
    __shared__ float sh[8];
    const long row = blockIdx.x;
    const float* xr = x + row * DM;
    float s = 0.f, s2 = 0.f;
    for (int d = threadIdx.x; d < DM; d += blockDim.x) {
        float v = xr[d]; s += v; s2 += v * v;
    }
    s  = blk_red(s,  sh, false);
    s2 = blk_red(s2, sh, false);
    const float mean = s / DM;
    const float var  = s2 / DM - mean * mean;
    const float rstd = rsqrtf(var + 1e-5f);
    half_t* orow = o + row * DM;
    for (int d = threadIdx.x; d < DM; d += blockDim.x)
        orow[d] = (half_t)((xr[d] - mean) * rstd * g[d] + b[d]);
}

// causal softmax per row (b,h,t); quirk preserved: scores = K@Q^T, no 1/sqrt(d)
__global__ __launch_bounds__(256) void softmax_k(float* __restrict__ w,
                                                 half_t* __restrict__ o16) {
    __shared__ float sh[8];
    const long row = blockIdx.x;
    const int  t = (int)(row % TT), valid = t + 1;
    float* wr = w + row * TT;
    float mx = -3.4e38f;
    for (int s = threadIdx.x; s < TT; s += blockDim.x)
        if (s < valid) mx = fmaxf(mx, wr[s]);
    mx = blk_red(mx, sh, true);
    float sum = 0.f;
    for (int s = threadIdx.x; s < TT; s += blockDim.x) {
        float e = (s < valid) ? __expf(wr[s] - mx) : 0.f;
        wr[s] = e; sum += e;
    }
    sum = blk_red(sum, sh, false);
    const float inv = 1.f / sum;
    half_t* orow = o16 + row * TT;
    for (int s = threadIdx.x; s < TT; s += blockDim.x)
        orow[s] = (half_t)(wr[s] * inv);
}

// attn16 [B,H,T,HD] -> attnT16 [B,T,H*HD]
__global__ void transpose_k(const half_t* __restrict__ a, half_t* __restrict__ o) {
    const int i = blockIdx.x * blockDim.x + threadIdx.x;
    if (i >= BT * DM) return;
    const int d = i % DM, bt = i / DM;
    const int b = bt / TT, t = bt % TT, h = d / HDI, e = d % HDI;
    o[i] = a[(((long)(b * NH + h)) * TT + t) * HDI + e];
}

__global__ __launch_bounds__(256) void nll_k(const float* __restrict__ lg,
                                             const int* __restrict__ tgt,
                                             float* __restrict__ nll) {
    __shared__ float sh[8];
    const long row = blockIdx.x;
    const float* lr = lg + row * (long)VOC;
    float mx = -3.4e38f;
    for (int j = threadIdx.x; j < VOC; j += blockDim.x) mx = fmaxf(mx, lr[j]);
    mx = blk_red(mx, sh, true);
    float sum = 0.f;
    for (int j = threadIdx.x; j < VOC; j += blockDim.x) sum += __expf(lr[j] - mx);
    sum = blk_red(sum, sh, false);
    if (threadIdx.x == 0) nll[row] = logf(sum) + mx - lr[tgt[row]];
}

__global__ __launch_bounds__(256) void nll_reduce_k(const float* __restrict__ nll,
                                                    float* __restrict__ loss) {
    __shared__ float sh[8];
    float s = 0.f;
    for (int i = threadIdx.x; i < BT; i += blockDim.x) s += nll[i];
    s = blk_red(s, sh, false);
    if (threadIdx.x == 0) *loss = s / (float)BT;
}

// ---------------------------------------------------------------------------
// Host side
// ---------------------------------------------------------------------------
static inline void gemm(hipStream_t st,
                        const half_t* A, long sAz, int aDiv, int ldA,
                        const half_t* W, long sWz, int wMod, int ldW,
                        const float* bias, long sBz, int bMod,
                        const float* Rsd, int ldR,
                        float* C32, long sC32z, int ldC32,
                        half_t* C16, long sC16z, int ldC16,
                        int M, int N, int K, int Z, int flags) {
    dim3 grid((N + TN - 1) / TN, (M + TM - 1) / TM, Z);
    wmma_gemm<<<grid, 256, 0, st>>>(A, sAz, aDiv, ldA, W, sWz, wMod, ldW,
                                    bias, sBz, bMod, Rsd, ldR,
                                    C32, sC32z, ldC32, C16, sC16z, ldC16,
                                    M, N, K, flags);
}

static inline char* wsalloc(char*& p, size_t bytes) {
    char* r = p;
    p += (bytes + 255) & ~(size_t)255;
    return r;
}

extern "C" void kernel_launch(void* const* d_in, const int* in_sizes, int n_in,
                              void* d_out, int out_size, void* d_ws, size_t ws_size,
                              hipStream_t stream) {
    (void)in_sizes; (void)n_in; (void)out_size; (void)ws_size;
    const int*   ids       = (const int*)  d_in[0];
    const int*   tgt       = (const int*)  d_in[1];
    const float* tok_emb   = (const float*)d_in[2];
    const float* pos_emb   = (const float*)d_in[3];
    const float* in_proj_w = (const float*)d_in[4];
    const float* in_proj_b = (const float*)d_in[5];
    const float* wk        = (const float*)d_in[6];
    const float* bk        = (const float*)d_in[7];
    const float* wq        = (const float*)d_in[8];
    const float* bq        = (const float*)d_in[9];
    const float* wv        = (const float*)d_in[10];
    const float* bv        = (const float*)d_in[11];
    const float* out_proj_w= (const float*)d_in[12];
    const float* out_proj_b= (const float*)d_in[13];
    const float* ff_w1     = (const float*)d_in[14];
    const float* ff_b1     = (const float*)d_in[15];
    const float* ff_w2     = (const float*)d_in[16];
    const float* ff_b2     = (const float*)d_in[17];
    const float* lna_g     = (const float*)d_in[18];
    const float* lna_b     = (const float*)d_in[19];
    const float* lnf_g     = (const float*)d_in[20];
    const float* lnf_b     = (const float*)d_in[21];
    const float* out_w     = (const float*)d_in[22];
    const float* out_b     = (const float*)d_in[23];

    float* logits = (float*)d_out;                 // [BT, VOC]
    float* loss   = logits + (size_t)BT * VOC;     // scalar

    char* wp = (char*)d_ws;
    const int Z = BB * NH;
    // f32 scratch
    float*  xbuf  = (float*) wsalloc(wp, (size_t)BT * DM * 4);
    float*  wei   = (float*) wsalloc(wp, (size_t)Z * TT * TT * 4);
    float*  nllb  = (float*) wsalloc(wp, (size_t)BT * 4);
    // f16 activations
    half_t* h16   = (half_t*)wsalloc(wp, (size_t)BT * DM * 2);
    half_t* h2_16 = (half_t*)wsalloc(wp, (size_t)BT * DM * 2);
    half_t* q16   = (half_t*)wsalloc(wp, (size_t)BT * DM * 2);
    half_t* k16   = (half_t*)wsalloc(wp, (size_t)BT * DM * 2);
    half_t* vT16  = (half_t*)wsalloc(wp, (size_t)BT * DM * 2);
    half_t* wei16 = (half_t*)wsalloc(wp, (size_t)Z * TT * TT * 2);
    half_t* at16  = (half_t*)wsalloc(wp, (size_t)BT * DM * 2);
    half_t* atT16 = (half_t*)wsalloc(wp, (size_t)BT * DM * 2);
    half_t* ffb16 = (half_t*)wsalloc(wp, (size_t)BT * FF * 2);
    half_t* x16   = (half_t*)wsalloc(wp, (size_t)BT * DM * 2);
    // f16 N-major (pre-transposed) weights
    half_t* ipT = (half_t*)wsalloc(wp, (size_t)NL * DM * DM * 2);
    half_t* wqT = (half_t*)wsalloc(wp, (size_t)NL * DM * DM * 2);
    half_t* wkT = (half_t*)wsalloc(wp, (size_t)NL * DM * DM * 2);
    half_t* wvT = (half_t*)wsalloc(wp, (size_t)NL * DM * DM * 2);
    half_t* opT = (half_t*)wsalloc(wp, (size_t)NL * DM * DM * 2);
    half_t* f1T = (half_t*)wsalloc(wp, (size_t)NL * DM * FF * 2);
    half_t* f2T = (half_t*)wsalloc(wp, (size_t)NL * DM * FF * 2);
    half_t* owT = (half_t*)wsalloc(wp, (size_t)VOC * DM * 2);

    auto cvtT = [&](const float* s, half_t* d, int K_, int N_, long Zb) {
        const long tot = Zb * (long)K_ * N_;
        cvtT16_k<<<(unsigned)((tot + 255) / 256), 256, 0, stream>>>(s, d, K_, N_, tot);
    };
    // weights -> f16, N-major, once per launch
    cvtT(in_proj_w,  ipT, DM, DM, NL);
    cvtT(wq,         wqT, DM, HDI, (long)NL * NH);
    cvtT(wk,         wkT, DM, HDI, (long)NL * NH);
    cvtT(wv,         wvT, DM, HDI, (long)NL * NH);
    cvtT(out_proj_w, opT, DM, DM, NL);
    cvtT(ff_w1,      f1T, DM, FF, NL);
    cvtT(ff_w2,      f2T, FF, DM, NL);
    cvtT(out_w,      owT, DM, VOC, 1);

    const int elt_blocks = (BT * DM + 255) / 256;
    embed_k<<<elt_blocks, 256, 0, stream>>>(ids, tok_emb, pos_emb, xbuf);

    const long sW_head = (long)HDI * DM;   // per-head weight stride (N-major)
    for (int l = 0; l < NL; ++l) {
        const long lDD = (long)l * DM * DM;
        // h16 = LN(x; lna)
        ln_k<<<BT, 256, 0, stream>>>(xbuf, lna_g + l * DM, lna_b + l * DM, h16);
        // h2 = h @ in_proj_w + b   (f16 out only)
        gemm(stream, h16, 0, 1, DM, ipT + lDD, 0, 1, DM,
             in_proj_b + l * DM, 0, 1, nullptr, 0,
             nullptr, 0, 0, h2_16, 0, DM, BT, DM, DM, 1, GF_BIAS | GF_OUT16);
        // per-head q,k (plain) and v (transposed [HD][T]) ; z = b*NH + h
        gemm(stream, h2_16, (long)TT * DM, NH, DM, wqT + lDD, sW_head, NH, DM,
             bq + l * DM, HDI, NH, nullptr, 0,
             nullptr, 0, 0, q16, (long)TT * HDI, HDI, TT, HDI, DM, Z, GF_BIAS | GF_OUT16);
        gemm(stream, h2_16, (long)TT * DM, NH, DM, wkT + lDD, sW_head, NH, DM,
             bk + l * DM, HDI, NH, nullptr, 0,
             nullptr, 0, 0, k16, (long)TT * HDI, HDI, TT, HDI, DM, Z, GF_BIAS | GF_OUT16);
        gemm(stream, h2_16, (long)TT * DM, NH, DM, wvT + lDD, sW_head, NH, DM,
             bv + l * DM, HDI, NH, nullptr, 0,
             nullptr, 0, 0, vT16, (long)HDI * TT, TT, TT, HDI, DM, Z, GF_BIAS | GF_OUT16T);
        // wei = k @ q^T (q16 [T][HD] is already the N-major B operand)
        gemm(stream, k16, (long)TT * HDI, 1, HDI, q16, (long)TT * HDI, Z, HDI,
             nullptr, 0, 1, nullptr, 0,
             wei, (long)TT * TT, TT, nullptr, 0, 0, TT, TT, HDI, Z, GF_OUT32);
        // causal softmax -> f16
        softmax_k<<<Z * TT, 256, 0, stream>>>(wei, wei16);
        // attn = wei @ v  (vT16 [HD][T] is the N-major B operand)
        gemm(stream, wei16, (long)TT * TT, 1, TT, vT16, (long)HDI * TT, Z, TT,
             nullptr, 0, 1, nullptr, 0,
             nullptr, 0, 0, at16, (long)TT * HDI, HDI, TT, HDI, TT, Z, GF_OUT16);
        // [B,H,T,HD] -> [B,T,D]
        transpose_k<<<elt_blocks, 256, 0, stream>>>(at16, atT16);
        // x = attnT @ out_proj_w + b + x  (f32 out)
        gemm(stream, atT16, 0, 1, DM, opT + lDD, 0, 1, DM,
             out_proj_b + l * DM, 0, 1, xbuf, DM,
             xbuf, 0, DM, nullptr, 0, 0, BT, DM, DM, 1, GF_BIAS | GF_RES | GF_OUT32);
        // h16 = LN(x; lnf)
        ln_k<<<BT, 256, 0, stream>>>(xbuf, lnf_g + l * DM, lnf_b + l * DM, h16);
        // ffb = relu(h @ ff_w1 + b)
        gemm(stream, h16, 0, 1, DM, f1T + (long)l * DM * FF, 0, 1, DM,
             ff_b1 + (long)l * FF, 0, 1, nullptr, 0,
             nullptr, 0, 0, ffb16, 0, FF, BT, FF, DM, 1, GF_BIAS | GF_RELU | GF_OUT16);
        // x = ffb @ ff_w2 + b + x  (f32 residual + f16 copy for logits)
        gemm(stream, ffb16, 0, 1, FF, f2T + (long)l * DM * FF, 0, 1, FF,
             ff_b2 + l * DM, 0, 1, xbuf, DM,
             xbuf, 0, DM, x16, 0, DM, BT, DM, FF, 1,
             GF_BIAS | GF_RES | GF_OUT32 | GF_OUT16);
    }

    // logits = x @ out_w + out_b
    gemm(stream, x16, 0, 1, DM, owT, 0, 1, DM,
         out_b, 0, 1, nullptr, 0,
         logits, 0, VOC, nullptr, 0, 0, BT, VOC, DM, 1, GF_BIAS | GF_OUT32);

    // loss
    nll_k<<<BT, 256, 0, stream>>>(logits, tgt, nllb);
    nll_reduce_k<<<1, 256, 0, stream>>>(nllb, loss);
}
